// StandardMoELayer_83691732730419
// MI455X (gfx1250) — matrix operand order repaired and checked by reading, not compiled, not run on hardware
//
#include <hip/hip_runtime.h>
#include <hip/hip_bf16.h>
#include <stdint.h>

#define NTOK 16384
#define DIM  1024
#define HID  4096
#define NEXP 8
#define RH   (HID/2)   // router hidden = 2048
#define LN_EPS 1e-5f

typedef __bf16 bf16_t;
typedef __bf16 v16bf __attribute__((ext_vector_type(16)));
typedef float  v8f   __attribute__((ext_vector_type(8)));

union FragAB { v16bf v; uint4 u[2]; };
union U4S    { uint4 u; unsigned short s[8]; };

__device__ inline bf16_t f2bf(float f) {
  union { float f; unsigned u; } in; in.f = f;
  unsigned u = in.u;
  u = (u + 0x7FFFu + ((u >> 16) & 1u)) >> 16;   // round-to-nearest-even
  union { unsigned short s; bf16_t b; } out; out.s = (unsigned short)u;
  return out.b;
}
__device__ inline float bf2f(bf16_t b) {
  union { unsigned short s; bf16_t b; } in; in.b = b;
  union { unsigned u; float f; } out; out.u = ((unsigned)in.s) << 16;
  return out.f;
}

// --- CDNA5 async global->LDS copy (ASYNCcnt-tracked DMA), 16B per lane ------
__device__ inline void async_ld_b128(unsigned lds_off, const void* gaddr) {
  asm volatile("global_load_async_to_lds_b128 %0, %1, off"
               :: "v"(lds_off), "v"(gaddr)
               : "memory");
}
__device__ inline void wait_async0() {
  asm volatile("s_wait_asynccnt 0x0" ::: "memory");
}

// ---------------------------------------------------------------------------
// f32 -> bf16 cast (weights), grid-stride
// ---------------------------------------------------------------------------
__global__ __launch_bounds__(256)
void moe_cast_f32_bf16(const float* __restrict__ src, bf16_t* __restrict__ dst,
                       long n) {
  long i = (long)blockIdx.x * 256 + threadIdx.x;
  long stride = (long)gridDim.x * 256;
  for (; i < n; i += stride) dst[i] = f2bf(src[i]);
}

// ---------------------------------------------------------------------------
// LayerNorm: one block per row (D=1024, 256 threads x 4 elems).
// Writes xn (bf16) and initializes out = x (residual).
// ---------------------------------------------------------------------------
__global__ __launch_bounds__(256)
void moe_layernorm(const float* __restrict__ x, const float* __restrict__ g,
                   const float* __restrict__ b, bf16_t* __restrict__ xn,
                   float* __restrict__ out) {
  int row = blockIdx.x;
  const float* xr = x + (long)row * DIM;
  __shared__ float s1[256], s2[256];
  float v[4]; float lsum = 0.f, lsq = 0.f;
#pragma unroll
  for (int i = 0; i < 4; ++i) {
    v[i] = xr[threadIdx.x + i * 256];
    lsum += v[i]; lsq += v[i] * v[i];
  }
  s1[threadIdx.x] = lsum; s2[threadIdx.x] = lsq;
  __syncthreads();
  for (int off = 128; off > 0; off >>= 1) {
    if (threadIdx.x < (unsigned)off) {
      s1[threadIdx.x] += s1[threadIdx.x + off];
      s2[threadIdx.x] += s2[threadIdx.x + off];
    }
    __syncthreads();
  }
  float mu  = s1[0] * (1.f / DIM);
  float var = s2[0] * (1.f / DIM) - mu * mu;
  float rstd = rsqrtf(var + LN_EPS);
#pragma unroll
  for (int i = 0; i < 4; ++i) {
    int c = threadIdx.x + i * 256;
    float nv = (v[i] - mu) * rstd * g[c] + b[c];
    xn[(long)row * DIM + c]  = f2bf(nv);
    out[(long)row * DIM + c] = v[i];          // residual init
  }
}

// ---------------------------------------------------------------------------
// Generic bf16 WMMA GEMM. Block tile 128x128, K-step 32, 256 threads = 8 waves,
// each wave a 64x32 tile via 8x v_wmma_f32_16x16x32_bf16.
// Double-buffered LDS pipeline:
//   A tile : global_load_async_to_lds_b128 DMA (no VGPR staging, ASYNCcnt)
//   B tile : register-staged, (k,k+1) bf16 pairs packed -> 8x ds_store_b32,
//            stored transposed [col][k] for contiguous fragment reads.
// One s_wait_asynccnt + barrier per k-step. Tail rows pre-zeroed in LDS once;
// their lanes never issue the async copy (EXEC predication).
//   GATHER_A : A rows indirected through gidx (token compaction; hoisted)
//   EPI == 0 : C = relu(A*B + bias) stored bf16 (row = slot)
//   EPI == 1 : out[gidx[row]] += gate[gidx[row]][gateCol] * (A*B + bias)
// M is dynamic via cntPtr (per-expert token count) when cntPtr != nullptr.
// ---------------------------------------------------------------------------
template<bool GATHER_A, int EPI>
__global__ __launch_bounds__(256, 1)
void moe_gemm_wmma(const bf16_t* __restrict__ A, const bf16_t* __restrict__ B,
                   const float* __restrict__ bias, bf16_t* __restrict__ Cb,
                   float* __restrict__ Cf, int M, const int* __restrict__ cntPtr,
                   int K, int Nn, const int* __restrict__ gidx,
                   const float* __restrict__ gate, int gateCol) {
  const int Meff = cntPtr ? *cntPtr : M;
  const int row0 = blockIdx.x * 128;
  if (row0 >= Meff) return;
  const int n0 = blockIdx.y * 128;
  const bool full = (row0 + 128 <= Meff);   // block-uniform fast path

  __shared__ __align__(16) bf16_t smA[2][128 * 40];  // [row][k], stride 40
  __shared__ __align__(16) bf16_t smB[2][128 * 40];  // transposed: [col][k]

  const int tid  = threadIdx.x;
  const int wid  = tid >> 5, lane = tid & 31;
  const int half = lane >> 4, l15 = lane & 15;
  const int mBase = (wid >> 2) * 64;                 // 0 / 64
  const int nBase = (wid & 3) * 32;                  // 0..96

  // ---- hoisted per-thread load descriptors (gather index resolved once) ----
  const bf16_t* aSrc[2]; bool aOk[2]; unsigned aLds[2];
#pragma unroll
  for (int it = 0; it < 2; ++it) {
    int i  = tid + it * 256;
    int r  = i >> 2;             // tile row 0..127
    int c8 = (i & 3) << 3;       // k-chunk 0/8/16/24
    int gr = row0 + r;
    aOk[it] = gr < Meff;
    long arow = aOk[it] ? (GATHER_A ? (long)gidx[gr] : (long)gr) : 0;
    aSrc[it] = A + arow * (long)K + c8;
    aLds[it] = (unsigned)(uintptr_t)(smA[0] + r * 40 + c8);  // buf0 LDS offset
    if (!full && !aOk[it]) {     // pre-zero tail rows in BOTH buffers, once
      uint4 z = {0u, 0u, 0u, 0u};
      *(uint4*)(smA[0] + r * 40 + c8) = z;
      *(uint4*)(smA[1] + r * 40 + c8) = z;
    }
  }
  const unsigned smABufStride = (unsigned)(128 * 40 * sizeof(bf16_t));

  // B: each thread owns k-rows {2*k2, 2*k2+1} at one 8-col chunk
  const int bK2 = (tid >> 4) << 1;          // even k row: 0,2,..,30
  const int bC8 = (tid & 15) << 3;          // col chunk
  const bf16_t* bSrc = B + (long)bK2 * Nn + n0 + bC8;

  const int nsteps = K >> 5;     // K / 32
  U4S stB0, stB1;

  // ---- A tile DMA: global -> LDS buffer `buf`, tile at k0 ----
  auto asyncA = [&](int buf, int k0) {
#pragma unroll
    for (int it = 0; it < 2; ++it)
      if (aOk[it])
        async_ld_b128(aLds[it] + (unsigned)buf * smABufStride,
                      (const void*)(aSrc[it] + k0));
  };
  // ---- B tile: global load into registers ----
  auto gloadB = [&](int k0) {
    const bf16_t* bp = bSrc + (long)k0 * Nn;
    stB0.u = *(const uint4*)(bp);
    stB1.u = *(const uint4*)(bp + Nn);
  };
  // ---- B tile: pack (k,k+1) pairs, store transposed to LDS buffer ----
  auto lstoreB = [&](int buf) {
#pragma unroll
    for (int j = 0; j < 8; ++j) {
      unsigned pk = (unsigned)stB0.s[j] | ((unsigned)stB1.s[j] << 16);
      *(unsigned*)(smB[buf] + (bC8 + j) * 40 + bK2) = pk;
    }
  };

  v8f acc[4][2];
#pragma unroll
  for (int mi = 0; mi < 4; ++mi)
#pragma unroll
    for (int ni = 0; ni < 2; ++ni) acc[mi][ni] = (v8f){};

  asyncA(0, 0);
  gloadB(0);
  lstoreB(0);
  wait_async0();
  __syncthreads();

  for (int s = 0; s < nsteps; ++s) {
    const int cur = s & 1;
    if (s + 1 < nsteps) {                      // next tile in flight during WMMAs
      asyncA(1 - cur, (s + 1) << 5);
      gloadB((s + 1) << 5);
    }

    FragAB fa[4], fb[2];
#pragma unroll
    for (int mi = 0; mi < 4; ++mi) {
      const bf16_t* ar = smA[cur] + (mBase + mi * 16 + l15) * 40 + half * 8;
      fa[mi].u[0] = *(const uint4*)(ar);        // K 0..7  (lanes>=16: 8..15)
      fa[mi].u[1] = *(const uint4*)(ar + 16);   // K 16..23 (lanes>=16: 24..31)
    }
#pragma unroll
    for (int ni = 0; ni < 2; ++ni) {
      const bf16_t* br = smB[cur] + (nBase + ni * 16 + l15) * 40 + half * 8;
      fb[ni].u[0] = *(const uint4*)(br);
      fb[ni].u[1] = *(const uint4*)(br + 16);
    }
#pragma unroll
    for (int mi = 0; mi < 4; ++mi)
#pragma unroll
      for (int ni = 0; ni < 2; ++ni)
        acc[mi][ni] = __builtin_amdgcn_wmma_f32_16x16x32_bf16(
            false, fa[mi].v, false, fb[ni].v, (short)0, acc[mi][ni],
            false, false);

    if (s + 1 < nsteps) {
      lstoreB(1 - cur);         // ds stores to the *other* buffer
      wait_async0();            // A DMA for next tile landed in LDS
      __syncthreads();          // single barrier per k-step
    }
  }

  // --- epilogue; C layout: lanes 0-15 -> M=j, lanes 16-31 -> M=j+8 ---
#pragma unroll
  for (int ni = 0; ni < 2; ++ni) {
    const int c = n0 + nBase + ni * 16 + l15;
    const float bv = bias[c];
#pragma unroll
    for (int mi = 0; mi < 4; ++mi) {
#pragma unroll
      for (int j = 0; j < 8; ++j) {
        int r = row0 + mBase + mi * 16 + half * 8 + j;
        if (!full && r >= Meff) continue;
        float val = acc[mi][ni][j] + bv;
        if (EPI == 0) {
          val = val > 0.f ? val : 0.f;
          Cb[(long)r * Nn + c] = f2bf(val);
        } else {
          int tok = gidx[r];
          float gv = gate[(long)tok * NEXP + gateCol];
          if (gv != 0.f) Cf[(long)tok * Nn + c] += gv * val;
        }
      }
    }
  }
}

// ---------------------------------------------------------------------------
// Router tail: logits = h_r @ r_w2 + b2, softmax(8), top-2, renormalize.
// Writes dense gates [N,8] and appends token to per-expert index lists.
// One block (256 threads) per token.
// ---------------------------------------------------------------------------
__global__ __launch_bounds__(256)
void moe_router(const bf16_t* __restrict__ hr, const float* __restrict__ w2,
                const float* __restrict__ b2, float* __restrict__ gate,
                int* __restrict__ cnt, int* __restrict__ eidx) {
  int t = blockIdx.x;
  const bf16_t* h = hr + (long)t * RH;
  float p[NEXP];
#pragma unroll
  for (int j = 0; j < NEXP; ++j) p[j] = 0.f;
  for (int i = threadIdx.x; i < RH; i += 256) {
    float hv = bf2f(h[i]);
    const float* wr = w2 + (long)i * NEXP;
#pragma unroll
    for (int j = 0; j < NEXP; ++j) p[j] += hv * wr[j];
  }
  __shared__ float sm[256 * NEXP];
#pragma unroll
  for (int j = 0; j < NEXP; ++j) sm[threadIdx.x * NEXP + j] = p[j];
  __syncthreads();
  for (int off = 128; off > 0; off >>= 1) {
    if (threadIdx.x < (unsigned)off)
#pragma unroll
      for (int j = 0; j < NEXP; ++j)
        sm[threadIdx.x * NEXP + j] += sm[(threadIdx.x + off) * NEXP + j];
    __syncthreads();
  }
  if (threadIdx.x == 0) {
    float l[NEXP]; float mx = -3.0e38f;
#pragma unroll
    for (int j = 0; j < NEXP; ++j) { l[j] = sm[j] + b2[j]; mx = fmaxf(mx, l[j]); }
    float sum = 0.f;
#pragma unroll
    for (int j = 0; j < NEXP; ++j) { l[j] = __expf(l[j] - mx); sum += l[j]; }
    float inv = 1.f / sum;
#pragma unroll
    for (int j = 0; j < NEXP; ++j) l[j] *= inv;   // probs
    int i1 = 0;
#pragma unroll
    for (int j = 1; j < NEXP; ++j) if (l[j] > l[i1]) i1 = j;
    int i2 = (i1 == 0) ? 1 : 0;
#pragma unroll
    for (int j = 0; j < NEXP; ++j) if (j != i1 && l[j] > l[i2]) i2 = j;
    float s = l[i1] + l[i2];
#pragma unroll
    for (int j = 0; j < NEXP; ++j) gate[(long)t * NEXP + j] = 0.f;
    gate[(long)t * NEXP + i1] = l[i1] / s;
    gate[(long)t * NEXP + i2] = l[i2] / s;
    int s1 = atomicAdd(&cnt[i1], 1); eidx[(long)i1 * NTOK + s1] = t;
    int s2 = atomicAdd(&cnt[i2], 1); eidx[(long)i2 * NTOK + s2] = t;
  }
}

__global__ void moe_zero_counts(int* cnt) {
  if (threadIdx.x < NEXP) cnt[threadIdx.x] = 0;
}

// ---------------------------------------------------------------------------
extern "C" void kernel_launch(void* const* d_in, const int* in_sizes, int n_in,
                              void* d_out, int out_size, void* d_ws,
                              size_t ws_size, hipStream_t stream) {
  (void)in_sizes; (void)n_in; (void)out_size; (void)ws_size;
  const float* x    = (const float*)d_in[0];
  const float* ln_g = (const float*)d_in[1];
  const float* ln_b = (const float*)d_in[2];
  const float* r_w1 = (const float*)d_in[3];
  const float* r_b1 = (const float*)d_in[4];
  const float* r_w2 = (const float*)d_in[5];
  const float* r_b2 = (const float*)d_in[6];
  const float* e_w1 = (const float*)d_in[7];
  const float* e_b1 = (const float*)d_in[8];
  const float* e_w2 = (const float*)d_in[9];
  const float* e_b2 = (const float*)d_in[10];
  float* out = (float*)d_out;

  // workspace carve (256B aligned)
  size_t off = 0;
  char* wsb = (char*)d_ws;
  auto carve = [&](size_t bytes) -> void* {
    void* p = wsb + off;
    off += (bytes + 255) & ~(size_t)255;
    return p;
  };
  bf16_t* xn   = (bf16_t*)carve((size_t)NTOK * DIM * 2);
  bf16_t* rw1b = (bf16_t*)carve((size_t)DIM * RH * 2);
  bf16_t* ew1b = (bf16_t*)carve((size_t)NEXP * DIM * HID * 2);
  bf16_t* ew2b = (bf16_t*)carve((size_t)NEXP * HID * DIM * 2);
  bf16_t* hr   = (bf16_t*)carve((size_t)NTOK * RH * 2);
  float*  gate = (float*)carve((size_t)NTOK * NEXP * 4);
  int*    cnt  = (int*)carve(256);
  int*    eidx = (int*)carve((size_t)NEXP * NTOK * 4);
  bf16_t* h1   = (bf16_t*)carve((size_t)NTOK * HID * 2);

  // 1) weight casts to bf16
  moe_cast_f32_bf16<<<2048, 256, 0, stream>>>(r_w1, rw1b, (long)DIM * RH);
  moe_cast_f32_bf16<<<8192, 256, 0, stream>>>(e_w1, ew1b, (long)NEXP * DIM * HID);
  moe_cast_f32_bf16<<<8192, 256, 0, stream>>>(e_w2, ew2b, (long)NEXP * HID * DIM);

  // 2) layernorm (+ residual init of out)
  moe_layernorm<<<NTOK, 256, 0, stream>>>(x, ln_g, ln_b, xn, out);

  // 3) router GEMM1: [N,1024] x [1024,2048], ReLU, bf16 out
  {
    dim3 g(NTOK / 128, RH / 128);
    moe_gemm_wmma<false, 0><<<g, 256, 0, stream>>>(
        xn, rw1b, r_b1, hr, nullptr, NTOK, nullptr, DIM, RH,
        nullptr, nullptr, 0);
  }

  // 4) router tail: logits + softmax + top-2 + token compaction
  moe_zero_counts<<<1, 32, 0, stream>>>(cnt);
  moe_router<<<NTOK, 256, 0, stream>>>(hr, r_w2, r_b2, gate, cnt, eidx);

  // 5) experts: gathered GEMM1 (ReLU -> H1), GEMM2 (scatter, gate-weighted)
  for (int e = 0; e < NEXP; ++e) {
    dim3 g1(NTOK / 128, HID / 128);
    moe_gemm_wmma<true, 0><<<g1, 256, 0, stream>>>(
        xn, ew1b + (size_t)e * DIM * HID, e_b1 + (size_t)e * HID,
        h1, nullptr, NTOK, cnt + e, DIM, HID,
        eidx + (size_t)e * NTOK, nullptr, 0);
    dim3 g2(NTOK / 128, DIM / 128);
    moe_gemm_wmma<false, 1><<<g2, 256, 0, stream>>>(
        h1, ew2b + (size_t)e * HID * DIM, e_b2 + (size_t)e * DIM,
        nullptr, out, NTOK, cnt + e, HID, DIM,
        eidx + (size_t)e * NTOK, gate, e);
  }
}
